// BinaryConnectConv_84421877170700
// MI455X (gfx1250) — compile-verified, matrix operand order checked by reading, not run on hardware
//
#include <hip/hip_runtime.h>
#include <cstdint>
#include <cstddef>

typedef __attribute__((ext_vector_type(16))) __bf16        v16bf;
typedef __attribute__((ext_vector_type(8)))  float         v8f;
typedef __attribute__((ext_vector_type(4)))  unsigned int  v4u;
typedef __attribute__((ext_vector_type(8)))  unsigned int  v8u;
typedef __attribute__((ext_vector_type(4)))  unsigned int  u32x4;
typedef __attribute__((ext_vector_type(8)))  int           i32x8;
typedef __attribute__((ext_vector_type(4)))  int           i32x4;

#define DIM_H   112
#define DIM_W   112
#define CIN     128
#define COUT    256
#define TAPS    9
#define NKB     4          // four K-blocks of 32 input channels per tap
#define LDSW    66         // 64-pixel tile + 2 halo columns
#define CSTR    136        // padded LDS channel stride (ushorts): 272B -> bank-conflict free
#define BPK_ELEMS (TAPS * NKB * 16 * 32 * 16)        // 294,912 bf16 weights
#define XBF_ELEMS ((size_t)32 * DIM_H * DIM_W * CIN) // 51,380,224 bf16 activations

// round-to-nearest-even fp32 -> bf16 (bit pattern)
__device__ __forceinline__ unsigned int bfbits(unsigned int u) {
    return (u + 0x7FFFu + ((u >> 16) & 1u)) >> 16;
}

// ---------------------------------------------------------------------------
// Kernel: binarize HWIO weights into the per-lane WMMA B-operand layout.
// Lane L<16  holds column N = nsub*16+L,      K = kb*32 + {0..7, 16..23}
// Lane L>=16 holds column N = nsub*16+(L-16), K = kb*32 + {8..15, 24..31}
// ---------------------------------------------------------------------------
__global__ void binconv_pack_weights(const float* __restrict__ kern,
                                     unsigned short* __restrict__ bpk) {
    int t = blockIdx.x * blockDim.x + threadIdx.x;
    if (t >= TAPS * NKB * 16 * 32) return;
    int lane = t & 31;
    int nsub = (t >> 5) & 15;
    int kb   = (t >> 9) & 3;
    int tap  = t >> 11;
    int n  = nsub * 16 + (lane & 15);
    int c0 = kb * 32 + ((lane & 16) ? 8 : 0);
    unsigned short* dst = bpk + (size_t)t * 16;
#pragma unroll
    for (int j = 0; j < 16; ++j) {
        int k = c0 + (j < 8 ? j : j + 8);
        float wv = kern[((size_t)tap * CIN + k) * COUT + n];
        dst[j] = (wv >= 0.0f) ? (unsigned short)0x3F80u : (unsigned short)0xBF80u;
    }
}

// ---------------------------------------------------------------------------
// Kernel: one-time fp32 -> bf16 conversion of the activations (8 floats/iter).
// The 103 MB bf16 tensor then lives in the 192 MB L2 for the conv pass.
// ---------------------------------------------------------------------------
__global__ void binconv_convert_x(const float* __restrict__ x,
                                  unsigned short* __restrict__ xbf,
                                  long long ngroups) {
    long long g = (long long)blockIdx.x * blockDim.x + threadIdx.x;
    long long stride = (long long)gridDim.x * blockDim.x;
    for (; g < ngroups; g += stride) {
        const v4u* src = reinterpret_cast<const v4u*>(x + (size_t)g * 8);
        v4u lo = src[0];
        v4u hi = src[1];
        v4u pk;
#pragma unroll
        for (int i = 0; i < 4; ++i) {
            unsigned int a = (i < 2) ? lo[2 * i]     : hi[2 * i - 4];
            unsigned int b = (i < 2) ? lo[2 * i + 1] : hi[2 * i - 3];
            pk[i] = bfbits(a) | (bfbits(b) << 16);
        }
        *reinterpret_cast<v4u*>(xbf + (size_t)g * 8) = pk;
    }
}

// ---------------------------------------------------------------------------
// Shared compute core: 8 waves, each owning 2 N-subtiles (32 channels),
// 4 M-subtiles, K = 9 taps x 128 ch  ->  288 v_wmma_f32_16x16x32_bf16 / wave.
// ---------------------------------------------------------------------------
__device__ __forceinline__ void binconv_compute_store(
        const unsigned short* ldsx, const unsigned short* __restrict__ bpk,
        float* __restrict__ out, int n, int h, int w0, int tid) {
    const int lane   = tid & 31;
    const int wave   = tid >> 5;
    const int lrow   = lane & 15;
    const int hiHalf = (lane >> 4) & 1;
    const int c0sel  = hiHalf * 8;

    v8f acc[4][2];
#pragma unroll
    for (int m = 0; m < 4; ++m)
#pragma unroll
        for (int b = 0; b < 2; ++b)
#pragma unroll
            for (int r = 0; r < 8; ++r) acc[m][b][r] = 0.0f;

#pragma unroll 1
    for (int tap = 0; tap < TAPS; ++tap) {
        const int ky = tap / 3;
        const int kx = tap % 3;
        if (tap + 1 < TAPS)
            __builtin_prefetch(bpk + ((size_t)(tap + 1) * NKB * 16 * 32) * 16, 0, 1);

#pragma unroll
        for (int kb = 0; kb < NKB; ++kb) {
            const unsigned short* bbase =
                bpk + ((((size_t)tap * NKB + kb) * 16 + wave * 2) * 32 + lane) * 16;
            v8u bu0 = *reinterpret_cast<const v8u*>(bbase);
            v8u bu1 = *reinterpret_cast<const v8u*>(bbase + 512);
            v16bf b0 = __builtin_bit_cast(v16bf, bu0);
            v16bf b1 = __builtin_bit_cast(v16bf, bu1);

            const int c0 = kb * 32 + c0sel;
#pragma unroll
            for (int m = 0; m < 4; ++m) {
                const unsigned short* ap =
                    &ldsx[(ky * LDSW + m * 16 + lrow + kx) * CSTR + c0];
                v4u alo = *reinterpret_cast<const v4u*>(ap);
                v4u ahi = *reinterpret_cast<const v4u*>(ap + 16);
                v8u au  = __builtin_shufflevector(alo, ahi, 0, 1, 2, 3, 4, 5, 6, 7);
                v16bf a = __builtin_bit_cast(v16bf, au);

                acc[m][0] = __builtin_amdgcn_wmma_f32_16x16x32_bf16(
                    false, a, false, b0, (short)0, acc[m][0], false, false);
                acc[m][1] = __builtin_amdgcn_wmma_f32_16x16x32_bf16(
                    false, a, false, b1, (short)0, acc[m][1], false, false);
            }
        }
    }

    const size_t rowbase = (((size_t)n * DIM_H + h) * DIM_W) * COUT;
    const int ch = wave * 32 + lrow;
#pragma unroll
    for (int m = 0; m < 4; ++m) {
#pragma unroll
        for (int r = 0; r < 8; ++r) {
            int wpix = w0 + m * 16 + hiHalf * 8 + r;
            if (wpix < DIM_W) {
                float* o = out + rowbase + (size_t)wpix * COUT + ch;
                o[0]  = acc[m][0][r];
                o[16] = acc[m][1][r];
            }
        }
    }
}

// ---------------------------------------------------------------------------
// Main kernel, TDM path: each block DMAs its 3 bf16 input rows into LDS with
// tensor_load_to_lds; the D# pad fields insert 8 ushorts per 128-ch pixel,
// reproducing the CSTR=136 bank-conflict-free LDS stride in hardware.
// ---------------------------------------------------------------------------
__launch_bounds__(256, 1)
__global__ void binconv_wmma_tdm_kernel(const unsigned short* __restrict__ xbf,
                                        const unsigned short* __restrict__ bpk,
                                        float* __restrict__ out) {
    __shared__ unsigned short ldsx[3 * LDSW * CSTR];   // group-segment offset 0

    const int w0  = blockIdx.x * 64;
    const int h   = blockIdx.y;
    const int n   = blockIdx.z;
    const int tid = threadIdx.x;
    const int wave = tid >> 5;

    // ---- zero halo columns / out-of-range rows (disjoint from TDM writes) ----
    for (int idx = tid; idx < 3 * LDSW; idx += 256) {
        int col = idx % LDSW;
        int ky  = idx / LDSW;
        int hin = h + ky - 1;
        int win = w0 + col - 1;
        if (!((unsigned)hin < DIM_H && (unsigned)win < DIM_W)) {
            v4u z = {0u, 0u, 0u, 0u};
            v4u* p = reinterpret_cast<v4u*>(&ldsx[(ky * LDSW + col) * CSTR]);
#pragma unroll
            for (int i = 0; i < 16; ++i) p[i] = z;     // 128 ushorts
        }
    }

    // ---- TDM: one 2D tensor_load_to_lds per valid input row (wave 0 only) ----
    if (wave == 0) {
        const int wlo   = (w0 == 0) ? 0 : (w0 - 1);
        const int whi   = (w0 + 65 < DIM_W) ? (w0 + 65) : DIM_W;
        const int ncols = whi - wlo;                   // clipped pixel count
        const int dcol  = wlo - (w0 - 1);              // LDS column of first pixel
#pragma unroll 1
        for (int ky = 0; ky < 3; ++ky) {
            int hin = h + ky - 1;
            if ((unsigned)hin >= DIM_H) continue;
            unsigned long long ga = (unsigned long long)(uintptr_t)(
                xbf + (((size_t)n * DIM_H + hin) * DIM_W + wlo) * CIN);
            unsigned lds_off = (unsigned)(((ky * LDSW + dcol) * CSTR) * 2);

            u32x4 g0;
            g0[0] = 1u;                                   // count=1, user D#
            g0[1] = lds_off;                              // lds_addr (bytes)
            g0[2] = (unsigned)(ga & 0xFFFFFFFFull);       // global_addr[31:0]
            g0[3] = (unsigned)((ga >> 32) & 0x01FFFFFFu)  // global_addr[56:32]
                  | (2u << 30);                           // type=2 ("image")

            // data_size=1(2B) | pad_enable | pad_interval=5(64 dw) | pad_amount=3(4 dw)
            i32x8 g1;
            g1[0] = (int)((1u << 16) | (1u << 20) | (5u << 22) | (3u << 25));
            g1[1] = (int)(((unsigned)CIN & 0xFFFFu) << 16);      // tensor_dim0 = 128
            g1[2] = (int)(((unsigned)ncols & 0xFFFFu) << 16);    // tensor_dim1 = ncols
            g1[3] = (int)(((unsigned)CIN & 0xFFFFu) << 16);      // tile_dim0 = 128
            g1[4] = (int)((unsigned)ncols & 0xFFFFu);            // tile_dim1 = ncols
            g1[5] = (int)(unsigned)CIN;                          // tensor_dim0_stride
            g1[6] = 0;
            g1[7] = 0;

            i32x4 z4 = {0, 0, 0, 0};
            i32x8 z8 = {0, 0, 0, 0, 0, 0, 0, 0};
            __builtin_amdgcn_tensor_load_to_lds(g0, g1, z4, z4, z8, 0);
        }
        __builtin_amdgcn_s_wait_tensorcnt(0);
    }
    __syncthreads();

    binconv_compute_store(ldsx, bpk, out, n, h, w0, tid);
}

// ---------------------------------------------------------------------------
// Fallback kernel (small workspace): manual fp32->bf16 staging into LDS.
// ---------------------------------------------------------------------------
__launch_bounds__(256, 1)
__global__ void binconv_wmma_kernel(const float* __restrict__ x,
                                    const unsigned short* __restrict__ bpk,
                                    float* __restrict__ out) {
    __shared__ unsigned short ldsx[3 * LDSW * CSTR];

    const int w0  = blockIdx.x * 64;
    const int h   = blockIdx.y;
    const int n   = blockIdx.z;
    const int tid = threadIdx.x;

    const int groups = 3 * LDSW * (CIN / 8);
    for (int g = tid; g < groups; g += 256) {
        int cg  = g & 15;
        int col = (g >> 4) % LDSW;
        int ky  = (g >> 4) / LDSW;
        int hin = h + ky - 1;
        int win = w0 + col - 1;
        v4u lo = {0u, 0u, 0u, 0u};
        v4u hi = {0u, 0u, 0u, 0u};
        if ((unsigned)hin < DIM_H && (unsigned)win < DIM_W) {
            const v4u* src = reinterpret_cast<const v4u*>(
                x + (((size_t)n * DIM_H + hin) * DIM_W + win) * CIN + cg * 8);
            lo = src[0];
            hi = src[1];
        }
        v4u pk;
#pragma unroll
        for (int i = 0; i < 4; ++i) {
            unsigned int a = (i < 2) ? lo[2 * i]     : hi[2 * i - 4];
            unsigned int b = (i < 2) ? lo[2 * i + 1] : hi[2 * i - 3];
            pk[i] = bfbits(a) | (bfbits(b) << 16);
        }
        *reinterpret_cast<v4u*>(&ldsx[(ky * LDSW + col) * CSTR + cg * 8]) = pk;
    }
    __syncthreads();

    binconv_compute_store(ldsx, bpk, out, n, h, w0, tid);
}

// ---------------------------------------------------------------------------
extern "C" void kernel_launch(void* const* d_in, const int* in_sizes, int n_in,
                              void* d_out, int out_size, void* d_ws, size_t ws_size,
                              hipStream_t stream) {
    (void)in_sizes; (void)n_in; (void)out_size;
    const float* x    = (const float*)d_in[0];
    const float* kern = (const float*)d_in[1];
    float* out        = (float*)d_out;

    unsigned short* bpk = (unsigned short*)d_ws;
    const size_t bpk_bytes = (size_t)BPK_ELEMS * 2;          // 589,824
    const size_t xbf_bytes = XBF_ELEMS * 2;                  // 102,760,448

    binconv_pack_weights<<<72, 256, 0, stream>>>(kern, bpk);

    dim3 grid(2, DIM_H, 32);
    if (ws_size >= bpk_bytes + xbf_bytes) {
        unsigned short* xbf = (unsigned short*)((char*)d_ws + bpk_bytes);
        long long ngroups = (long long)(XBF_ELEMS / 8);
        binconv_convert_x<<<4096, 256, 0, stream>>>(x, xbf, ngroups);
        binconv_wmma_tdm_kernel<<<grid, 256, 0, stream>>>(xbf, bpk, out);
    } else {
        binconv_wmma_kernel<<<grid, 256, 0, stream>>>(x, bpk, out);
    }
}
